// MultiHeadAttention_61057255080010
// MI455X (gfx1250) — matrix operand order repaired and checked by reading, not compiled, not run on hardware
//
#include <hip/hip_runtime.h>
#include <hip/hip_bf16.h>
#include <stdint.h>

// Problem constants (from reference)
namespace {
constexpr int kB = 4, kT = 2048, kC = 2048, kH = 16, kD = 128;
constexpr int kNQKV = 3 * kC;     // 6144
constexpr int kRows = kB * kT;    // 8192 tokens
}

typedef __attribute__((ext_vector_type(16))) __bf16 v16bf;
typedef __attribute__((ext_vector_type(8)))  __bf16 v8bf;
typedef __attribute__((ext_vector_type(8)))  float  v8f;

typedef __attribute__((ext_vector_type(4))) int          i32x4;
typedef __attribute__((ext_vector_type(8))) int          i32x8;
typedef __attribute__((ext_vector_type(4))) unsigned int u32x4;

// Address-space qualified int4 for the async-to-LDS builtin (param type per
// hipcc diagnostic: pointer to 16-byte int vector; global side first).
typedef __attribute__((address_space(1))) i32x4 i32x4_as1;
typedef __attribute__((address_space(3))) i32x4 i32x4_as3;

#if __has_builtin(__builtin_amdgcn_global_load_async_to_lds_b128)
#define HAVE_ASYNC_LDS 1
#else
#define HAVE_ASYNC_LDS 0
#endif
#if __has_builtin(__builtin_amdgcn_tensor_load_to_lds)
#define HAVE_TDM 1
#else
#define HAVE_TDM 0
#endif

// Flat LDS addresses keep the byte offset in the low 32 bits (ISA 10.2).
static __device__ __forceinline__ unsigned lds_byte_addr(const void* p) {
  return (unsigned)(uintptr_t)p;
}

// One b128 per lane, memory -> LDS, tracked by ASYNCcnt (no data VGPRs).
static __device__ __forceinline__ void async_b128(const void* g, void* l) {
#if HAVE_ASYNC_LDS
  __builtin_amdgcn_global_load_async_to_lds_b128(
      (i32x4_as1*)(uintptr_t)g, (i32x4_as3*)(uintptr_t)l, 0, 0);
#else
  *(uint4*)l = *(const uint4*)g;
#endif
}

static __device__ __forceinline__ void wait_async_all() {
#if HAVE_ASYNC_LDS
#if __has_builtin(__builtin_amdgcn_s_wait_asynccnt)
  __builtin_amdgcn_s_wait_asynccnt(0);
#else
  asm volatile("s_wait_asynccnt 0" ::: "memory");
#endif
#endif
}

static __device__ __forceinline__ void wait_tensor() {
#if __has_builtin(__builtin_amdgcn_s_wait_tensorcnt)
  __builtin_amdgcn_s_wait_tensorcnt(0);
#else
  asm volatile("s_wait_tensorcnt 0" ::: "memory");
#endif
}

#if HAVE_TDM
// Tensor Data Mover: 2D tile (tile_dw dwords x tile_rows rows) from global to
// LDS.  Descriptor bit layout per CDNA5 ISA §8.3/8.4 (count=1, type=2,
// data_size=4B units, 2D: remaining groups zero).  Tracked by TENSORcnt.
// 6-arg builtin arity: (u32x4, i32x8, i32x4, i32x4, i32x8, i32 cpol).
static __device__ __forceinline__ void tdm_load_2d(
    unsigned lds_byte, const void* gptr, unsigned tile_dw, unsigned tile_rows,
    unsigned stride_dw, unsigned tensor_rows) {
  const unsigned long long ga = (unsigned long long)(uintptr_t)gptr;
  u32x4 g0;
  g0[0] = 1u;                                                // desc count = 1
  g0[1] = lds_byte;                                          // lds_addr
  g0[2] = (unsigned)ga;                                      // global_addr lo
  g0[3] = (unsigned)((ga >> 32) & 0x01FFFFFFu) | (2u << 30); // addr hi | type=2
  i32x8 g1;
  g1[0] = (int)(2u << 16);                                   // data_size = 4B
  g1[1] = (int)((stride_dw & 0xFFFFu) << 16);                // tensor_dim0 lo16
  g1[2] = (int)((stride_dw >> 16) | ((tensor_rows & 0xFFFFu) << 16));
  g1[3] = (int)((tensor_rows >> 16) | (tile_dw << 16));      // tile_dim0
  g1[4] = (int)tile_rows;                                    // tile_dim1
  g1[5] = (int)stride_dw;                                    // dim0_stride lo32
  g1[6] = 0;
  g1[7] = 0;
  i32x4 z4 = {};
  i32x8 z8 = {};
  __builtin_amdgcn_tensor_load_to_lds(g0, g1, z4, z4, z8, 0);
}
#endif

// Build a 16-element bf16 fragment from two 16-byte chunks.
static __device__ __forceinline__ v16bf pack16(const __bf16* lo, const __bf16* hi) {
  v8bf a = *(const v8bf*)lo;
  v8bf b = *(const v8bf*)hi;
  v16bf r;
#pragma unroll
  for (int i = 0; i < 8; ++i) { r[i] = a[i]; r[8 + i] = b[i]; }
  return r;
}

static __device__ __forceinline__ v8f wmma_bf16(v16bf a, v16bf b, v8f c) {
  // D(16x16,f32) = A(16x32,bf16) * B(32x16,bf16) + C
  return __builtin_amdgcn_wmma_f32_16x16x32_bf16(false, a, false, b, (short)0, c,
                                                 false, false);
}

// ---------------------------------------------------------------------------
// Weight transpose + fp32 -> bf16.  in: [K][N] f32, out: [N][K] bf16.
// ---------------------------------------------------------------------------
__global__ __launch_bounds__(256) void transpose_cvt_kernel(
    const float* __restrict__ in, __bf16* __restrict__ out, int K, int N) {
  __shared__ float tile[32][33];
  const int n0 = blockIdx.x * 32, k0 = blockIdx.y * 32;
  const int tx = threadIdx.x, ty = threadIdx.y;
#pragma unroll
  for (int i = ty; i < 32; i += 8)
    tile[i][tx] = in[(size_t)(k0 + i) * N + n0 + tx];
  __syncthreads();
#pragma unroll
  for (int i = ty; i < 32; i += 8)
    out[(size_t)(n0 + i) * K + k0 + tx] = (__bf16)tile[tx][i];
}

// ---------------------------------------------------------------------------
// QKV GEMM: [8192,2048]f32 x [2048,6144] -> scatter q,k:[B,H,T,D], v:[B,H,D,T]
// Block tile 128x128, K-step 32, double-buffered LDS, 8 waves (4 M x 2 N).
// A staged with fp32->bf16 conversion; B tile staged by the Tensor Data Mover
// (one 2D descriptor per tile, wave 0 issues, TENSORcnt-waited).
// ---------------------------------------------------------------------------
__global__ __launch_bounds__(256) void qkv_gemm_kernel(
    const float* __restrict__ x, const __bf16* __restrict__ wT,
    __bf16* __restrict__ qo, __bf16* __restrict__ ko, __bf16* __restrict__ vt) {
  __shared__ alignas(16) __bf16 As[2][128][32];
  __shared__ alignas(16) __bf16 Bs[2][128][32];
  const int tid = threadIdx.x;
  const int r0 = blockIdx.x * 128;   // token rows
  const int n0 = blockIdx.y * 128;   // qkv columns
  const int wave = tid >> 5, lane = tid & 31;
  const int wm = wave & 3, wn = wave >> 2;
  const int row = lane & 15, lh = lane >> 4;   // fragment row / K-half select
  const int srow = tid >> 1, sseg = (tid & 1) * 16;  // staging: row, 16-elem seg

  v8f acc[2][4] = {};

  auto stageA = [&](int kt, int buf) {
    const int k0 = kt * 32;
    const float* pa = x + (size_t)(r0 + srow) * kC + k0 + sseg;
    v8bf t0, t1;
#pragma unroll
    for (int j = 0; j < 2; ++j) {
      float4 f = *(const float4*)(pa + 4 * j);
      t0[4 * j + 0] = (__bf16)f.x; t0[4 * j + 1] = (__bf16)f.y;
      t0[4 * j + 2] = (__bf16)f.z; t0[4 * j + 3] = (__bf16)f.w;
    }
#pragma unroll
    for (int j = 0; j < 2; ++j) {
      float4 f = *(const float4*)(pa + 8 + 4 * j);
      t1[4 * j + 0] = (__bf16)f.x; t1[4 * j + 1] = (__bf16)f.y;
      t1[4 * j + 2] = (__bf16)f.z; t1[4 * j + 3] = (__bf16)f.w;
    }
    *(v8bf*)&As[buf][srow][sseg]     = t0;
    *(v8bf*)&As[buf][srow][sseg + 8] = t1;
    __builtin_prefetch(pa + 32, 0, 1);
  };

#if HAVE_TDM
  auto stageB = [&](int kt, int buf) {   // whole 128x32 bf16 tile, one TDM op
    const int k0 = kt * 32;
    tdm_load_2d(lds_byte_addr(&Bs[buf][0][0]),
                wT + (size_t)n0 * kC + k0,
                /*tile_dw=*/16, /*tile_rows=*/128,
                /*stride_dw=*/kC / 2, /*tensor_rows=*/(unsigned)kNQKV);
  };
#else
  auto stageB = [&](int kt, int buf) {
    const int k0 = kt * 32;
    const __bf16* pb = wT + (size_t)(n0 + srow) * kC + k0 + sseg;
    async_b128(pb,     &Bs[buf][srow][sseg]);
    async_b128(pb + 8, &Bs[buf][srow][sseg + 8]);
    __builtin_prefetch(pb + 32, 0, 1);
  };
#endif

  stageA(0, 0);
#if HAVE_TDM
  if (wave == 0) stageB(0, 0);
#else
  stageB(0, 0);
#endif
  const int KT = kC / 32;
  for (int kt = 0; kt < KT; ++kt) {
    const int buf = kt & 1;
#if HAVE_TDM
    if (wave == 0) wait_tensor();
#else
    wait_async_all();
#endif
    __syncthreads();
    if (kt + 1 < KT) {
      stageA(kt + 1, buf ^ 1);
#if HAVE_TDM
      if (wave == 0) stageB(kt + 1, buf ^ 1);
#else
      stageB(kt + 1, buf ^ 1);
#endif
    }
    // A fragments: row = lane%16, elems = K[lh*8 .. +7] and K[lh*8+16 .. +23]
    v16bf a0 = pack16(&As[buf][wm * 32 + row][lh * 8],
                      &As[buf][wm * 32 + row][lh * 8 + 16]);
    v16bf a1 = pack16(&As[buf][wm * 32 + 16 + row][lh * 8],
                      &As[buf][wm * 32 + 16 + row][lh * 8 + 16]);
#pragma unroll
    for (int nt = 0; nt < 4; ++nt) {
      // B fragment: col = lane%16, elems = K[lh*8 .. +7] and K[lh*8+16 .. +23]
      const __bf16* pb = &Bs[buf][wn * 64 + nt * 16 + row][lh * 8];
      v16bf bfrag = pack16(pb, pb + 16);
      acc[0][nt] = wmma_bf16(a0, bfrag, acc[0][nt]);
      acc[1][nt] = wmma_bf16(a1, bfrag, acc[1][nt]);
    }
  }
  // Epilogue: scatter into q/k [B,H,T,D] and v transposed [B,H,D,T] (bf16)
#pragma unroll
  for (int mt = 0; mt < 2; ++mt)
#pragma unroll
    for (int nt = 0; nt < 4; ++nt)
#pragma unroll
      for (int e = 0; e < 8; ++e) {
        const int gr = r0 + wm * 32 + mt * 16 + e + 8 * lh;  // token row
        const int gn = n0 + wn * 64 + nt * 16 + row;          // qkv col
        const float val = acc[mt][nt][e];
        const int part = gn >> 11;        // 0:q 1:k 2:v
        const int cc = gn & 2047;
        const int h = cc >> 7, d = cc & 127;
        const int bb = gr >> 11, t = gr & 2047;
        const size_t hb = (size_t)(bb * kH + h);
        if (part == 0)      qo[(hb * kT + t) * kD + d] = (__bf16)val;
        else if (part == 1) ko[(hb * kT + t) * kD + d] = (__bf16)val;
        else                vt[(hb * kD + d) * kT + t] = (__bf16)val;
      }
}

// ---------------------------------------------------------------------------
// RoPE (fp32 math) applied in place to q,k [B,H,T,D] bf16.
// ---------------------------------------------------------------------------
__global__ __launch_bounds__(256) void rope_kernel(__bf16* __restrict__ q,
                                                   __bf16* __restrict__ k) {
  const int gid = blockIdx.x * 256 + threadIdx.x;   // B*H*T*64 threads
  const int i = gid & 63;
  const int rowi = gid >> 6;
  const int t = rowi & (kT - 1);
  const size_t base = (size_t)rowi * kD;
  const float freq = (float)t * __powf(10000.f, -(float)i * (1.0f / 64.0f));
  float s, c;
  __sincosf(freq, &s, &c);
  {
    float x1 = (float)q[base + i], x2 = (float)q[base + i + 64];
    q[base + i]      = (__bf16)(x1 * c - x2 * s);
    q[base + i + 64] = (__bf16)(x2 * c + x1 * s);
  }
  {
    float x1 = (float)k[base + i], x2 = (float)k[base + i + 64];
    k[base + i]      = (__bf16)(x1 * c - x2 * s);
    k[base + i + 64] = (__bf16)(x2 * c + x1 * s);
  }
}

// ---------------------------------------------------------------------------
// Flash attention: 4 waves/block, each wave owns 16 query rows x D=128.
// Streams 32-key blocks: 8 WMMA for S = Q*K^T, online softmax, 8 WMMA for P*V.
// q,k:[B,H,T,D] bf16; vt:[B,H,D,T] bf16; ao:[B,H,T,D] bf16.
// ---------------------------------------------------------------------------
__global__ __launch_bounds__(128) void attn_kernel(
    const __bf16* __restrict__ q, const __bf16* __restrict__ k,
    const __bf16* __restrict__ vt, __bf16* __restrict__ ao) {
  __shared__ alignas(16) __bf16 P[4][16][32];  // per-wave C->A layout shuffle
  const int wave = threadIdx.x >> 5;
  const int lane = threadIdx.x & 31;
  const int row = lane & 15, lh = lane >> 4;
  const int bh = blockIdx.x >> 5;              // B*H heads, T/64 qblocks each
  const int qblk = blockIdx.x & 31;
  const int q0 = qblk * 64 + wave * 16;
  const __bf16* qp = q  + (size_t)bh * kT * kD;
  const __bf16* kp = k  + (size_t)bh * kT * kD;
  const __bf16* vp = vt + (size_t)bh * kD * kT;

  // Q fragments for the wave's 16 rows, K-dim 128 in 4 chunks of 32.
  v16bf aq[4];
#pragma unroll
  for (int kc = 0; kc < 4; ++kc) {
    const __bf16* p = qp + (size_t)(q0 + row) * kD + kc * 32 + lh * 16;
    aq[kc] = pack16(p, p + 8);
  }

  v8f o[8] = {};
  float mr[8], lr[8];
#pragma unroll
  for (int e = 0; e < 8; ++e) { mr[e] = -1e30f; lr[e] = 0.f; }

  const float scale = 0.08838834764831845f;  // 1/sqrt(128)
  const int qend = q0 + 15;
  for (int kb = 0; kb <= qend; kb += 32) {
    v8f s0 = {}, s1 = {};
#pragma unroll
    for (int kc = 0; kc < 4; ++kc) {
      // B = K^T fragment: col = key (lane%16), elems along d (contiguous)
      const __bf16* p0 = kp + (size_t)(kb + row) * kD + kc * 32 + lh * 16;
      const __bf16* p1 = kp + (size_t)(kb + 16 + row) * kD + kc * 32 + lh * 16;
      s0 = wmma_bf16(aq[kc], pack16(p0, p0 + 8), s0);
      s1 = wmma_bf16(aq[kc], pack16(p1, p1 + 8), s1);
    }
    const bool need_mask = (kb + 31 > q0);
    float bm[8], rs[8], alpha[8];
#pragma unroll
    for (int e = 0; e < 8; ++e) {
      float a = s0[e] * scale;
      float b = s1[e] * scale;
      if (need_mask) {
        const int qr = q0 + e + 8 * lh;
        if (kb + row > qr)      a = -1e30f;
        if (kb + 16 + row > qr) b = -1e30f;
      }
      s0[e] = a; s1[e] = b;
      bm[e] = fmaxf(a, b);
    }
#pragma unroll
    for (int off = 1; off < 16; off <<= 1)
#pragma unroll
      for (int e = 0; e < 8; ++e)
        bm[e] = fmaxf(bm[e], __shfl_xor(bm[e], off, 32));
#pragma unroll
    for (int e = 0; e < 8; ++e) {
      const float mnew = fmaxf(mr[e], bm[e]);
      alpha[e] = __expf(mr[e] - mnew);
      mr[e] = mnew;
      const float p0 = __expf(s0[e] - mnew);
      const float p1 = __expf(s1[e] - mnew);
      s0[e] = p0; s1[e] = p1;
      rs[e] = p0 + p1;
    }
#pragma unroll
    for (int off = 1; off < 16; off <<= 1)
#pragma unroll
      for (int e = 0; e < 8; ++e)
        rs[e] += __shfl_xor(rs[e], off, 32);
#pragma unroll
    for (int e = 0; e < 8; ++e) lr[e] = lr[e] * alpha[e] + rs[e];
#pragma unroll
    for (int dt = 0; dt < 8; ++dt)
#pragma unroll
      for (int e = 0; e < 8; ++e) o[dt][e] *= alpha[e];
    // C-layout -> A-layout via wave-private LDS tile (bf16)
#pragma unroll
    for (int e = 0; e < 8; ++e) {
      P[wave][e + 8 * lh][row]      = (__bf16)s0[e];
      P[wave][e + 8 * lh][16 + row] = (__bf16)s1[e];
    }
    v16bf ap = pack16(&P[wave][row][lh * 16], &P[wave][row][lh * 16 + 8]);
#pragma unroll
    for (int dt = 0; dt < 8; ++dt) {
      // B = V fragment: col = d (lane%16), elems along key (contiguous in vt)
      const __bf16* pv = vp + (size_t)(dt * 16 + row) * kT + kb + lh * 16;
      o[dt] = wmma_bf16(ap, pack16(pv, pv + 8), o[dt]);
    }
  }
  float inv[8];
#pragma unroll
  for (int e = 0; e < 8; ++e) inv[e] = 1.0f / lr[e];
#pragma unroll
  for (int dt = 0; dt < 8; ++dt)
#pragma unroll
    for (int e = 0; e < 8; ++e) {
      const size_t idx = ((size_t)bh * kT + q0 + e + 8 * lh) * kD + dt * 16 + row;
      ao[idx] = (__bf16)(o[dt][e] * inv[e]);
    }
}

// ---------------------------------------------------------------------------
// Output projection: ao[B,H,T,D]bf16 (gathered as [8192,2048]) x wT + bias.
// Both A and B tiles staged with GLOBAL_LOAD_ASYNC_TO_LDS (ASYNCcnt).
// ---------------------------------------------------------------------------
__global__ __launch_bounds__(256) void proj_gemm_kernel(
    const __bf16* __restrict__ ao, const __bf16* __restrict__ wT,
    const float* __restrict__ bias, float* __restrict__ out) {
  __shared__ alignas(16) __bf16 As[2][128][32];
  __shared__ alignas(16) __bf16 Bs[2][128][32];
  const int tid = threadIdx.x;
  const int r0 = blockIdx.x * 128;
  const int n0 = blockIdx.y * 128;
  const int wave = tid >> 5, lane = tid & 31;
  const int wm = wave & 3, wn = wave >> 2;
  const int row = lane & 15, lh = lane >> 4;
  const int srow = tid >> 1, sseg = (tid & 1) * 16;

  v8f acc[2][4] = {};

  auto stage = [&](int kt, int buf) {
    const int k0 = kt * 32;
    const int c0 = k0 + sseg;            // channel = h*128 + d
    const int h = c0 >> 7, dd = c0 & 127;
    const int r = r0 + srow;
    const int bb = r >> 11, t = r & 2047;
    const __bf16* pa = ao + ((size_t)(bb * kH + h) * kT + t) * kD + dd;
    async_b128(pa,     &As[buf][srow][sseg]);
    async_b128(pa + 8, &As[buf][srow][sseg + 8]);
    const __bf16* pb = wT + (size_t)(n0 + srow) * kC + c0;
    async_b128(pb,     &Bs[buf][srow][sseg]);
    async_b128(pb + 8, &Bs[buf][srow][sseg + 8]);
    __builtin_prefetch(pb + 32, 0, 1);
  };

  stage(0, 0);
  const int KT = kC / 32;
  for (int kt = 0; kt < KT; ++kt) {
    const int buf = kt & 1;
    wait_async_all();
    __syncthreads();
    if (kt + 1 < KT) stage(kt + 1, buf ^ 1);
    v16bf a0 = pack16(&As[buf][wm * 32 + row][lh * 8],
                      &As[buf][wm * 32 + row][lh * 8 + 16]);
    v16bf a1 = pack16(&As[buf][wm * 32 + 16 + row][lh * 8],
                      &As[buf][wm * 32 + 16 + row][lh * 8 + 16]);
#pragma unroll
    for (int nt = 0; nt < 4; ++nt) {
      const __bf16* pb = &Bs[buf][wn * 64 + nt * 16 + row][lh * 8];
      v16bf bfrag = pack16(pb, pb + 16);
      acc[0][nt] = wmma_bf16(a0, bfrag, acc[0][nt]);
      acc[1][nt] = wmma_bf16(a1, bfrag, acc[1][nt]);
    }
  }
#pragma unroll
  for (int mt = 0; mt < 2; ++mt)
#pragma unroll
    for (int nt = 0; nt < 4; ++nt)
#pragma unroll
      for (int e = 0; e < 8; ++e) {
        const int gr = r0 + wm * 32 + mt * 16 + e + 8 * lh;
        const int gn = n0 + wn * 64 + nt * 16 + row;
        out[(size_t)gr * kC + gn] = acc[mt][nt][e] + bias[gn];
      }
}

// ---------------------------------------------------------------------------
extern "C" void kernel_launch(void* const* d_in, const int* in_sizes, int n_in,
                              void* d_out, int out_size, void* d_ws,
                              size_t ws_size, hipStream_t stream) {
  (void)in_sizes; (void)n_in; (void)out_size; (void)ws_size;
  const float* x      = (const float*)d_in[0];
  const float* w_qkv  = (const float*)d_in[1];
  const float* w_proj = (const float*)d_in[2];
  const float* b_proj = (const float*)d_in[3];
  float* out = (float*)d_out;

  char* ws = (char*)d_ws;
  size_t off = 0;
  __bf16* wqkvT  = (__bf16*)(ws + off); off += (size_t)kNQKV * kC * 2;        // 24 MiB
  __bf16* wprojT = (__bf16*)(ws + off); off += (size_t)kC * kC * 2;           //  8 MiB
  __bf16* qbuf   = (__bf16*)(ws + off); off += (size_t)kRows * kC * 2;        // 32 MiB
  __bf16* kbuf   = (__bf16*)(ws + off); off += (size_t)kRows * kC * 2;        // 32 MiB
  __bf16* vtbuf  = (__bf16*)(ws + off); off += (size_t)kRows * kC * 2;        // 32 MiB
  __bf16* aobuf  = (__bf16*)(ws + off); off += (size_t)kRows * kC * 2;        // 32 MiB

  dim3 tpb32x8(32, 8);
  transpose_cvt_kernel<<<dim3(kNQKV / 32, kC / 32), tpb32x8, 0, stream>>>(
      w_qkv, wqkvT, kC, kNQKV);
  transpose_cvt_kernel<<<dim3(kC / 32, kC / 32), tpb32x8, 0, stream>>>(
      w_proj, wprojT, kC, kC);
  qkv_gemm_kernel<<<dim3(kRows / 128, kNQKV / 128), 256, 0, stream>>>(
      x, wqkvT, qbuf, kbuf, vtbuf);
  rope_kernel<<<(kB * kH * kT * 64) / 256, 256, 0, stream>>>(qbuf, kbuf);
  attn_kernel<<<kB * kH * (kT / 64), 128, 0, stream>>>(qbuf, kbuf, vtbuf, aobuf);
  proj_gemm_kernel<<<dim3(kRows / 128, kC / 128), 256, 0, stream>>>(
      aobuf, wprojT, b_proj, out);
}
